// ManModel_59983513255985
// MI455X (gfx1250) — matrix operand order repaired and checked by reading, not compiled, not run on hardware
//
#include <hip/hip_runtime.h>

#define BATCH 256
#define SEQL 256
#define HISL 1024
#define EMB 128
#define HID 256
#define POI 20000
#define CATN 400
#define TVN 48
#define NH 4
#define DK 64

typedef __bf16 bf16_t;
typedef bf16_t v16bf __attribute__((ext_vector_type(16)));
typedef bf16_t v8bf  __attribute__((ext_vector_type(8)));
typedef float  v8f   __attribute__((ext_vector_type(8)));

// ---------------- WMMA helpers (CDNA5 16x16x32 bf16) ----------------

__device__ __forceinline__ v8f zero8() {
    v8f z = {0.f,0.f,0.f,0.f,0.f,0.f,0.f,0.f};
    return z;
}

__device__ __forceinline__ v8f wmma_bf(v16bf a, v16bf b, v8f c) {
    return __builtin_amdgcn_wmma_f32_16x16x32_bf16(false, a, false, b, (short)0, c, false, false);
}

// A fragment: 16x32 bf16 tile, row-major with leading dim ld.
// lane L: row = L%16; k-chunks at half*8 and 16+half*8 (per ISA A layout).
__device__ __forceinline__ v16bf load_a(const bf16_t* p, int ld) {
    int lane = threadIdx.x & 31;
    int row = lane & 15, half = lane >> 4;
    const bf16_t* q = p + (size_t)row * ld + half * 8;
    v8bf c0 = *(const v8bf*)(q);
    v8bf c1 = *(const v8bf*)(q + 16);
    return __builtin_shufflevector(c0, c1, 0,1,2,3,4,5,6,7,8,9,10,11,12,13,14,15);
}

// B fragment: 32x16 bf16 tile from row-major [K,N]; lane L holds column L%16, k-chunks at half*8 and 16+half*8.
__device__ __forceinline__ v16bf load_b(const bf16_t* p, int ld) {
    int lane = threadIdx.x & 31;
    const bf16_t* q = p + (lane & 15) + (size_t)((lane >> 4) * 8) * ld;
    v16bf b;
#pragma unroll
    for (int j = 0; j < 8; j++) { b[j] = q[(size_t)j * ld]; b[8 + j] = q[(size_t)(16 + j) * ld]; }
    return b;
}

// row reductions within 16-lane halves (C-layout rows live on one half)
__device__ __forceinline__ float rsum16(float v) {
    v += __shfl_xor(v, 1, 32); v += __shfl_xor(v, 2, 32);
    v += __shfl_xor(v, 4, 32); v += __shfl_xor(v, 8, 32);
    return v;
}
__device__ __forceinline__ float rmax16(float v) {
    v = fmaxf(v, __shfl_xor(v, 1, 32)); v = fmaxf(v, __shfl_xor(v, 2, 32));
    v = fmaxf(v, __shfl_xor(v, 4, 32)); v = fmaxf(v, __shfl_xor(v, 8, 32));
    return v;
}

__device__ __forceinline__ float eluf(float v) { return v > 0.f ? v : __expf(v) - 1.f; }

// ---------------- kernel: transpose-convert weight [N,K] f32 -> [K,N] bf16 ----------------

__global__ __launch_bounds__(256) void pack_t(const float* __restrict__ w,
                                              bf16_t* __restrict__ out, int N, int K) {
    size_t i = (size_t)blockIdx.x * blockDim.x + threadIdx.x;
    if (i >= (size_t)N * K) return;
    int k = (int)(i / N), n = (int)(i % N);
    out[i] = (bf16_t)w[(size_t)n * K + k];
}

__global__ __launch_bounds__(256) void zero_f32(float* p, int n) {
    int i = blockIdx.x * blockDim.x + threadIdx.x;
    if (i < n) p[i] = 0.f;
}

// ---------------- kernel: tab = tanh(onehot @ Wt + b), N=128, K=POI ----------------

__global__ __launch_bounds__(32) void tab_gemm(const float* __restrict__ onehot,
                                               const bf16_t* __restrict__ Wt,
                                               const float* __restrict__ bias,
                                               bf16_t* __restrict__ tab) {
    int mt = blockIdx.x * 16;
    int lane = threadIdx.x & 31;
    int row = lane & 15, half = lane >> 4, c15 = lane & 15;
    v8f acc[8];
#pragma unroll
    for (int f = 0; f < 8; f++) acc[f] = zero8();
    for (int k0 = 0; k0 < POI; k0 += 32) {
        const float* ap = onehot + (size_t)(mt + row) * POI + k0 + half * 8;
        if (k0 + 32 < POI) {
            __builtin_prefetch(ap + 32, 0, 1);                        // next A tile
            __builtin_prefetch(Wt + (size_t)(k0 + 32 + lane) * 128, 0, 1); // next B rows
        }
        v16bf a;
#pragma unroll
        for (int j = 0; j < 8; j++) a[j] = (bf16_t)ap[j];
#pragma unroll
        for (int j = 0; j < 8; j++) a[8 + j] = (bf16_t)ap[16 + j];
#pragma unroll
        for (int f = 0; f < 8; f++)
            acc[f] = wmma_bf(a, load_b(Wt + (size_t)k0 * 128 + f * 16, 128), acc[f]);
    }
#pragma unroll
    for (int f = 0; f < 8; f++)
#pragma unroll
        for (int r = 0; r < 8; r++) {
            int col = f * 16 + c15;
            int rm = mt + r + 8 * half;
            tab[(size_t)rm * 128 + col] = (bf16_t)tanhf(acc[f][r] + bias[col]);
        }
}

// ---------------- kernel: fused embedding (gather + 2 MLP GEMMs + elu + PE) ----------------

__global__ __launch_bounds__(64) void embed_kernel(
    const int* __restrict__ traj, const int* __restrict__ timev, const int* __restrict__ catv,
    const float* __restrict__ poi_table, const float* __restrict__ poi_pin,
    const bf16_t* __restrict__ time_tab, const bf16_t* __restrict__ cat_tab,
    const bf16_t* __restrict__ main_Wt, const float* __restrict__ main_b,
    const bf16_t* __restrict__ other_Wt, const float* __restrict__ other_b,
    const float* __restrict__ freq_W, bf16_t* __restrict__ xout, int seqLen) {
    __shared__ __align__(16) bf16_t meL[16 * 128];
    __shared__ __align__(16) bf16_t oiL[16 * 256];
    __shared__ float w0s[16], w1s[16];
    __shared__ int trs[16], tms[16], cts[16];

    int tile = blockIdx.x, tid = threadIdx.x;
    if (tid < 16) {
        int t = tile * 16 + tid;
        int tr = traj[t];
        float u = 1.f - poi_pin[tr];
        float fw0 = freq_W[0], fw1 = freq_W[1];
        w0s[tid] = sqrtf(1.f / (1.f + __expf(-u * fw0)));
        w1s[tid] = sqrtf(1.f / (1.f + __expf(-u * fw1)));
        trs[tid] = tr; tms[tid] = timev[t]; cts[tid] = catv[t];
    }
    __syncthreads();
    for (int i = tid; i < 16 * 128; i += 64) {
        int tk = i >> 7, c = i & 127;
        meL[i] = (bf16_t)poi_table[(size_t)trs[tk] * 128 + c];
    }
    for (int i = tid; i < 16 * 256; i += 64) {
        int tk = i >> 8, c = i & 255;
        float v;
        if (c < 128) v = w0s[tk] * (float)cat_tab[(size_t)cts[tk] * 128 + c];
        else         v = w1s[tk] * (float)time_tab[(size_t)tms[tk] * 128 + (c - 128)];
        oiL[i] = (bf16_t)v;
    }
    __syncthreads();

    int wave = tid >> 5, lane = tid & 31, c15 = lane & 15, half = lane >> 4;
    int n_base = wave * 128;
    v8f am[8], ao[8];
#pragma unroll
    for (int f = 0; f < 8; f++) { am[f] = zero8(); ao[f] = zero8(); }
    for (int k0 = 0; k0 < 128; k0 += 32) {
        v16bf a = load_a(meL + k0, 128);
#pragma unroll
        for (int f = 0; f < 8; f++)
            am[f] = wmma_bf(a, load_b(main_Wt + (size_t)k0 * 256 + n_base + f * 16, 256), am[f]);
    }
    for (int k0 = 0; k0 < 256; k0 += 32) {
        v16bf a = load_a(oiL + k0, 256);
#pragma unroll
        for (int f = 0; f < 8; f++)
            ao[f] = wmma_bf(a, load_b(other_Wt + (size_t)k0 * 256 + n_base + f * 16, 256), ao[f]);
    }
#pragma unroll
    for (int f = 0; f < 8; f++)
#pragma unroll
        for (int r = 0; r < 8; r++) {
            int col = n_base + f * 16 + c15;
            int token = tile * 16 + r + 8 * half;
            float x = eluf(am[f][r] + main_b[col]) + eluf(ao[f][r] + other_b[col]);
            int pos = token & (seqLen - 1);
            int i2 = col & ~1;
            float im = __powf(10000.f, (float)i2 * (1.f / 256.f));
            float ang = (float)pos / im;
            x += (col & 1) ? __cosf(ang) : __sinf(ang);
            xout[(size_t)token * 256 + col] = (bf16_t)x;
        }
}

// ---------------- kernel: QKV projection (bf16 GEMM + head-split store) ----------------

__global__ __launch_bounds__(256) void gemm_qkv(const bf16_t* __restrict__ X,
                                                const bf16_t* __restrict__ Wt,
                                                const float* __restrict__ bias,
                                                bf16_t* __restrict__ out) {
    int wid = blockIdx.x * 8 + (threadIdx.x >> 5); // tasks = 4096 Mtiles * 4 Ngroups
    int mt = (wid >> 2) * 16;
    int n0 = (wid & 3) * 64;
    int lane = threadIdx.x & 31, c15 = lane & 15, half = lane >> 4;
    v8f acc[4];
#pragma unroll
    for (int j = 0; j < 4; j++) acc[j] = zero8();
    for (int k0 = 0; k0 < 256; k0 += 32) {
        v16bf a = load_a(X + (size_t)mt * 256 + k0, 256);
#pragma unroll
        for (int j = 0; j < 4; j++)
            acc[j] = wmma_bf(a, load_b(Wt + (size_t)k0 * 256 + n0 + j * 16, 256), acc[j]);
    }
#pragma unroll
    for (int j = 0; j < 4; j++)
#pragma unroll
        for (int r = 0; r < 8; r++) {
            int col = n0 + j * 16 + c15;
            int row = mt + r + 8 * half;
            int h = col >> 6, d = col & 63, b = row >> 8, l = row & 255;
            out[(((size_t)b * NH + h) * SEQL + l) * DK + d] = (bf16_t)(acc[j][r] + bias[col]);
        }
}

// ---------------- kernel: fused self-attention per (b,h); head0 probs -> d_out ----------------

__global__ __launch_bounds__(128) void attn_kernel(const bf16_t* __restrict__ Q,
                                                   const bf16_t* __restrict__ Kb,
                                                   const bf16_t* __restrict__ Vb,
                                                   const int* __restrict__ traj,
                                                   bf16_t* __restrict__ ctx,
                                                   float* __restrict__ attn0) {
    extern __shared__ __align__(16) char smem[];
    bf16_t* KT    = (bf16_t*)smem;                       // [64][256] K^T
    bf16_t* Pb    = (bf16_t*)(smem + 64 * 256 * 2);      // [4 waves][16][256]
    float*  maskv = (float*)(smem + 64 * 256 * 2 + 4 * 16 * 256 * 2);

    int bh = blockIdx.x, bb = bh >> 2, h = bh & 3;
    int tid = threadIdx.x, wave = tid >> 5, lane = tid & 31, c15 = lane & 15, half = lane >> 4;
    size_t base = ((size_t)bb * NH + h) * SEQL * DK;

    for (int i = tid; i < DK * SEQL; i += 128) {
        int d = i >> 8, l = i & 255;
        KT[d * 256 + l] = Kb[base + (size_t)l * DK + d];
    }
    for (int i = tid; i < SEQL; i += 128)
        maskv[i] = (traj[bb * SEQL + i] > 0) ? 0.f : -1.0e9f;
    // warm L2/WGP$ for V while S-phase runs
    __builtin_prefetch(Vb + base + (size_t)tid * 128, 0, 1);
    __syncthreads();

    bf16_t* Pw = Pb + wave * 16 * 256;
    for (int it = 0; it < 4; it++) {
        int m0 = wave * 64 + it * 16;
        v8f acc[16];
#pragma unroll
        for (int f = 0; f < 16; f++) acc[f] = zero8();
        for (int d0 = 0; d0 < 64; d0 += 32) {
            v16bf a = load_a(Q + base + (size_t)m0 * DK + d0, DK);
#pragma unroll
            for (int f = 0; f < 16; f++)
                acc[f] = wmma_bf(a, load_b(KT + d0 * 256 + f * 16, 256), acc[f]);
        }
#pragma unroll
        for (int r = 0; r < 8; r++) {
            float vals[16];
            float mx = -3.0e38f;
#pragma unroll
            for (int f = 0; f < 16; f++) {
                float v = acc[f][r] * 0.125f + maskv[f * 16 + c15];
                vals[f] = v; mx = fmaxf(mx, v);
            }
            mx = rmax16(mx);
            float s = 0.f;
#pragma unroll
            for (int f = 0; f < 16; f++) { float e = __expf(vals[f] - mx); vals[f] = e; s += e; }
            s = rsum16(s);
            float inv = 1.f / s;
            int rl = r + 8 * half;
            int row = m0 + rl;
#pragma unroll
            for (int f = 0; f < 16; f++) {
                float p = vals[f] * inv;
                Pw[rl * 256 + f * 16 + c15] = (bf16_t)p;
                if (h == 0) attn0[((size_t)bb * SEQL + row) * SEQL + f * 16 + c15] = p;
            }
        }
        __threadfence_block(); // order P writes before LDS A-frag reads (same wave)
        v8f acc2[4];
#pragma unroll
        for (int j = 0; j < 4; j++) acc2[j] = zero8();
        for (int k0 = 0; k0 < 256; k0 += 32) {
            v16bf a = load_a(Pw + k0, 256);
#pragma unroll
            for (int j = 0; j < 4; j++)
                acc2[j] = wmma_bf(a, load_b(Vb + base + (size_t)k0 * DK + j * 16, DK), acc2[j]);
        }
#pragma unroll
        for (int j = 0; j < 4; j++)
#pragma unroll
            for (int r = 0; r < 8; r++) {
                int row = m0 + r + 8 * half;
                int d = j * 16 + c15;
                ctx[((size_t)bb * SEQL + row) * HID + h * DK + d] = (bf16_t)acc2[j][r];
            }
    }
}

// ---------------- kernel: sa_out = ctx@sWo + b, LN, mask, partial mean -> poi_sum ----------------

__global__ __launch_bounds__(128) void swo_ln_kernel(const bf16_t* __restrict__ ctxb,
                                                     const bf16_t* __restrict__ sWo_t,
                                                     const float* __restrict__ sbo,
                                                     const float* __restrict__ sn_g,
                                                     const float* __restrict__ sn_b,
                                                     const int* __restrict__ traj,
                                                     float* __restrict__ poi_sum) {
    int wave = threadIdx.x >> 5;
    int mt = blockIdx.x * 64 + wave * 16;
    int lane = threadIdx.x & 31, c15 = lane & 15, half = lane >> 4;
    v8f acc[16];
#pragma unroll
    for (int f = 0; f < 16; f++) acc[f] = zero8();
    const bf16_t* Arow = ctxb + (size_t)mt * 256;
    for (int k0 = 0; k0 < 256; k0 += 32) {
        v16bf a = load_a(Arow + k0, 256);
#pragma unroll
        for (int f = 0; f < 16; f++)
            acc[f] = wmma_bf(a, load_b(sWo_t + (size_t)k0 * 256 + f * 16, 256), acc[f]);
    }
#pragma unroll
    for (int r = 0; r < 8; r++) {
        float s = 0.f, qq = 0.f;
#pragma unroll
        for (int f = 0; f < 16; f++) {
            float v = acc[f][r] + sbo[f * 16 + c15];
            acc[f][r] = v; s += v; qq += v * v;
        }
        s = rsum16(s); qq = rsum16(qq);
        float mean = s * (1.f / 256.f);
        float var = qq * (1.f / 256.f) - mean * mean;
        float inv = rsqrtf(var + 1e-5f);
        int row = mt + r + 8 * half;
        float mk = (traj[row] > 0) ? 1.f : 0.f;
#pragma unroll
        for (int f = 0; f < 16; f++) {
            int col = f * 16 + c15;
            acc[f][r] = ((acc[f][r] - mean) * inv * sn_g[col] + sn_b[col]) * mk;
        }
    }
    int bbb = mt >> 8;
#pragma unroll
    for (int f = 0; f < 16; f++) {
        float cs = 0.f;
#pragma unroll
        for (int r = 0; r < 8; r++) cs += acc[f][r];
        cs += __shfl_xor(cs, 16, 32);
        if (half == 0) atomicAdd(&poi_sum[bbb * 256 + f * 16 + c15], cs);
    }
}

// ---------------- kernel: poi_hidden = poi_sum/L ; aq = poi_hidden @ aWq^T ----------------

__global__ __launch_bounds__(256) void finalize_kernel(const float* __restrict__ poi_sum,
                                                       const float* __restrict__ aWq,
                                                       float* __restrict__ poi_hidden,
                                                       float* __restrict__ aqout,
                                                       bf16_t* __restrict__ seqh) {
    int bb = blockIdx.x, t = threadIdx.x;
    __shared__ float s[256];
    float ph = poi_sum[bb * 256 + t] * (1.f / 256.f);
    poi_hidden[bb * 256 + t] = ph;
    seqh[(size_t)bb * 512 + t] = (bf16_t)ph;
    s[t] = ph;
    __syncthreads();
    float acc = 0.f;
    for (int k = 0; k < 256; k++) acc += s[k] * aWq[(size_t)t * 256 + k];
    aqout[bb * 256 + t] = acc;
}

// ---------------- kernel: fused additive history attention per batch row ----------------

__global__ __launch_bounds__(256) void addattn_kernel(const bf16_t* __restrict__ his_x,
                                                      const bf16_t* __restrict__ aWk_t,
                                                      const bf16_t* __restrict__ aWv_t,
                                                      const float* __restrict__ aWa,
                                                      const float* __restrict__ aWo,
                                                      const float* __restrict__ ln_g,
                                                      const float* __restrict__ ln_b,
                                                      const float* __restrict__ aq,
                                                      const float* __restrict__ poi_hidden,
                                                      const int* __restrict__ his_traj,
                                                      bf16_t* __restrict__ seqh) {
    int bb = blockIdx.x, tid = threadIdx.x;
    int lane = tid & 31, wave = tid >> 5, c15 = lane & 15, half = lane >> 4;
    __shared__ float aqs[256], aws[256], actx[256];
    __shared__ float asc[1024];
    __shared__ float red[16];
    aqs[tid] = aq[bb * 256 + tid];
    aws[tid] = aWa[tid];
    actx[tid] = 0.f;
    __syncthreads();

    // pass 1: asc[k] = tanh(aq + his@aWk^T) . aWa   (+ mask)
    for (int tile = wave; tile < 64; tile += 8) {
        int t0 = tile * 16;
        v8f acc[16];
#pragma unroll
        for (int f = 0; f < 16; f++) acc[f] = zero8();
        const bf16_t* Arow = his_x + ((size_t)bb * HISL + t0) * 256;
        if (tile + 8 < 64)  // prefetch this wave's next 16x256 his_x tile (128 rows ahead)
            __builtin_prefetch(Arow + (size_t)128 * 256 + (size_t)lane * 128, 0, 1);
        for (int k0 = 0; k0 < 256; k0 += 32) {
            v16bf a = load_a(Arow + k0, 256);
#pragma unroll
            for (int f = 0; f < 16; f++)
                acc[f] = wmma_bf(a, load_b(aWk_t + (size_t)k0 * 256 + f * 16, 256), acc[f]);
        }
        float pr[8];
#pragma unroll
        for (int r = 0; r < 8; r++) pr[r] = 0.f;
#pragma unroll
        for (int f = 0; f < 16; f++) {
            int col = f * 16 + c15;
            float aqv = aqs[col], wa = aws[col];
#pragma unroll
            for (int r = 0; r < 8; r++) pr[r] += tanhf(aqv + acc[f][r]) * wa;
        }
#pragma unroll
        for (int r = 0; r < 8; r++) pr[r] = rsum16(pr[r]);
        if (c15 == 0) {
#pragma unroll
            for (int r = 0; r < 8; r++) {
                int row = t0 + r + 8 * half;
                float hm = (his_traj[(size_t)bb * HISL + row] > 0) ? 0.f : -1.0e9f;
                asc[row] = pr[r] + hm;
            }
        }
    }
    __syncthreads();

    // softmax over 1024
    float lm = -3.0e38f;
    for (int i = tid; i < 1024; i += 256) lm = fmaxf(lm, asc[i]);
#pragma unroll
    for (int m = 1; m < 32; m <<= 1) lm = fmaxf(lm, __shfl_xor(lm, m, 32));
    if (lane == 0) red[wave] = lm;
    __syncthreads();
    float bm = red[0];
#pragma unroll
    for (int i = 1; i < 8; i++) bm = fmaxf(bm, red[i]);
    __syncthreads();
    float ls = 0.f;
    for (int i = tid; i < 1024; i += 256) { float e = __expf(asc[i] - bm); asc[i] = e; ls += e; }
#pragma unroll
    for (int m = 1; m < 32; m <<= 1) ls += __shfl_xor(ls, m, 32);
    if (lane == 0) red[8 + wave] = ls;
    __syncthreads();
    float bs = 0.f;
#pragma unroll
    for (int i = 0; i < 8; i++) bs += red[8 + i];
    float inv = 1.f / bs;
    __syncthreads();
    for (int i = tid; i < 1024; i += 256) asc[i] *= inv;
    __syncthreads();

    // pass 2: actx = sum_k aw[k] * (his@aWv^T)[k]
    for (int tile = wave; tile < 64; tile += 8) {
        int t0 = tile * 16;
        v8f acc[16];
#pragma unroll
        for (int f = 0; f < 16; f++) acc[f] = zero8();
        const bf16_t* Arow = his_x + ((size_t)bb * HISL + t0) * 256;
        if (tile + 8 < 64)
            __builtin_prefetch(Arow + (size_t)128 * 256 + (size_t)lane * 128, 0, 1);
        for (int k0 = 0; k0 < 256; k0 += 32) {
            v16bf a = load_a(Arow + k0, 256);
#pragma unroll
            for (int f = 0; f < 16; f++)
                acc[f] = wmma_bf(a, load_b(aWv_t + (size_t)k0 * 256 + f * 16, 256), acc[f]);
        }
#pragma unroll
        for (int f = 0; f < 16; f++) {
            float cs = 0.f;
#pragma unroll
            for (int r = 0; r < 8; r++) cs += asc[t0 + r + 8 * half] * acc[f][r];
            cs += __shfl_xor(cs, 16, 32);
            if (half == 0) atomicAdd(&actx[f * 16 + c15], cs);
        }
    }
    __syncthreads();

    // add_out = actx @ aWo^T ; LN ; his_hidden = poi_hidden + LN
    float ao = 0.f;
    for (int n = 0; n < 256; n++) ao += actx[n] * aWo[(size_t)tid * 256 + n];
    float ssum = ao, sq = ao * ao;
#pragma unroll
    for (int m = 1; m < 32; m <<= 1) { ssum += __shfl_xor(ssum, m, 32); sq += __shfl_xor(sq, m, 32); }
    if (lane == 0) { red[wave] = ssum; red[8 + wave] = sq; }
    __syncthreads();
    float ts = 0.f, tq = 0.f;
#pragma unroll
    for (int i = 0; i < 8; i++) { ts += red[i]; tq += red[8 + i]; }
    float mean = ts * (1.f / 256.f);
    float var = tq * (1.f / 256.f) - mean * mean;
    float y = (ao - mean) * rsqrtf(var + 1e-5f) * ln_g[tid] + ln_b[tid];
    float hh = poi_hidden[bb * 256 + tid] + y;
    seqh[(size_t)bb * 512 + 256 + tid] = (bf16_t)hh;
}

// ---------------- kernel: logits = seq_hidden[256,512] @ Wt[512,N] + bias (f32 out) ----------------

__global__ __launch_bounds__(256) void logit_gemm(const bf16_t* __restrict__ seqh,
                                                  const bf16_t* __restrict__ Wt,
                                                  const float* __restrict__ bias,
                                                  float* __restrict__ out, int Ntiles) {
    int wid = blockIdx.x * 8 + (threadIdx.x >> 5);
    if (wid >= 16 * Ntiles) return;
    int mt = (wid / Ntiles) * 16;
    int n0 = (wid % Ntiles) * 16;
    int N = Ntiles * 16;
    int lane = threadIdx.x & 31, c15 = lane & 15, half = lane >> 4;
    v8f acc = zero8();
    for (int k0 = 0; k0 < 512; k0 += 32) {
        if (k0 + 32 < 512)  // prefetch next B K-rows of the streamed weight
            __builtin_prefetch(Wt + (size_t)(k0 + 32 + lane) * N + n0, 0, 1);
        v16bf a = load_a(seqh + (size_t)mt * 512 + k0, 512);
        v16bf b = load_b(Wt + (size_t)k0 * N + n0, N);
        acc = wmma_bf(a, b, acc);
    }
#pragma unroll
    for (int r = 0; r < 8; r++) {
        int row = mt + r + 8 * half, col = n0 + c15;
        out[(size_t)row * N + col] = acc[r] + bias[col];
    }
}

// ---------------- kernel: top-10 of logit_cat per row (indices as float) ----------------

__global__ __launch_bounds__(32) void topk_cat(const float* __restrict__ lc, float* __restrict__ outp) {
    int bb = blockIdx.x, lane = threadIdx.x;
    const float* row = lc + (size_t)bb * CATN;
    int sel[10];
#pragma unroll
    for (int it = 0; it < 10; it++) {
        float bv = -3.0e38f; int bi = 0x7fffffff;
        for (int i = lane; i < CATN; i += 32) {
            bool used = false;
#pragma unroll
            for (int j = 0; j < 10; j++) if (j < it && sel[j] == i) used = true;
            if (!used) {
                float v = row[i];
                if (v > bv || (v == bv && i < bi)) { bv = v; bi = i; }
            }
        }
#pragma unroll
        for (int m = 1; m < 32; m <<= 1) {
            float ov = __shfl_xor(bv, m, 32);
            int oi = __shfl_xor(bi, m, 32);
            if (ov > bv || (ov == bv && oi < bi)) { bv = ov; bi = oi; }
        }
        sel[it] = bi;
    }
    if (lane < 10) outp[bb * 10 + lane] = (float)sel[lane];
}

// ---------------- host ----------------

extern "C" void kernel_launch(void* const* d_in, const int* in_sizes, int n_in,
                              void* d_out, int out_size, void* d_ws, size_t ws_size,
                              hipStream_t stream) {
    const int* traj      = (const int*)d_in[0];
    const int* timev     = (const int*)d_in[1];
    const int* catv      = (const int*)d_in[2];
    const int* his_traj  = (const int*)d_in[5];
    const int* his_time  = (const int*)d_in[6];
    const int* his_cat   = (const int*)d_in[7];
    const float* tp_one  = (const float*)d_in[9];
    const float* ct_one  = (const float*)d_in[10];
    const float* W_time  = (const float*)d_in[11];
    const float* b_time  = (const float*)d_in[12];
    const float* W_cat   = (const float*)d_in[13];
    const float* b_cat   = (const float*)d_in[14];
    const float* poi_tab = (const float*)d_in[15];
    const float* poi_pin = (const float*)d_in[16];
    const float* main_W  = (const float*)d_in[18];
    const float* main_b  = (const float*)d_in[19];
    const float* other_W = (const float*)d_in[20];
    const float* other_b = (const float*)d_in[21];
    const float* freq_W  = (const float*)d_in[22];
    const float* sWq = (const float*)d_in[23]; const float* sbq = (const float*)d_in[24];
    const float* sWk = (const float*)d_in[25]; const float* sbk = (const float*)d_in[26];
    const float* sWv = (const float*)d_in[27]; const float* sbv = (const float*)d_in[28];
    const float* sWo = (const float*)d_in[29]; const float* sbo = (const float*)d_in[30];
    const float* sn_g = (const float*)d_in[31]; const float* sn_b = (const float*)d_in[32];
    const float* ln_g = (const float*)d_in[33]; const float* ln_b = (const float*)d_in[34];
    const float* aWq = (const float*)d_in[35];
    const float* aWk = (const float*)d_in[36];
    const float* aWv = (const float*)d_in[37];
    const float* aWo = (const float*)d_in[38];
    const float* aWa = (const float*)d_in[39];
    const float* poi_W = (const float*)d_in[40]; const float* poi_b = (const float*)d_in[41];
    const float* cat_W = (const float*)d_in[42]; const float* cat_b = (const float*)d_in[43];

    float* out_poi  = (float*)d_out;
    float* out_cat  = out_poi + (size_t)BATCH * POI;
    float* out_attn = out_cat + (size_t)BATCH * CATN;
    float* out_pred = out_attn + (size_t)BATCH * SEQL * SEQL;

    // workspace bump allocator (256B aligned)
    char* wp = (char*)d_ws;
    auto alloc = [&](size_t bytes) -> void* {
        void* p = (void*)wp;
        wp += (bytes + 255) & ~(size_t)255;
        return p;
    };
    bf16_t* main_Wt = (bf16_t*)alloc((size_t)128 * 256 * 2);
    bf16_t* other_Wt= (bf16_t*)alloc((size_t)256 * 256 * 2);
    bf16_t* sWq_t   = (bf16_t*)alloc((size_t)256 * 256 * 2);
    bf16_t* sWk_t   = (bf16_t*)alloc((size_t)256 * 256 * 2);
    bf16_t* sWv_t   = (bf16_t*)alloc((size_t)256 * 256 * 2);
    bf16_t* sWo_t   = (bf16_t*)alloc((size_t)256 * 256 * 2);
    bf16_t* aWk_t   = (bf16_t*)alloc((size_t)256 * 256 * 2);
    bf16_t* aWv_t   = (bf16_t*)alloc((size_t)256 * 256 * 2);
    bf16_t* Wtime_t = (bf16_t*)alloc((size_t)POI * 128 * 2);
    bf16_t* Wcat_t  = (bf16_t*)alloc((size_t)POI * 128 * 2);
    bf16_t* poi_Wt  = (bf16_t*)alloc((size_t)512 * POI * 2);
    bf16_t* cat_Wt  = (bf16_t*)alloc((size_t)512 * CATN * 2);
    bf16_t* time_tab= (bf16_t*)alloc((size_t)TVN * 128 * 2);
    bf16_t* cat_tab = (bf16_t*)alloc((size_t)CATN * 128 * 2);
    bf16_t* seq_x   = (bf16_t*)alloc((size_t)BATCH * SEQL * 256 * 2);
    bf16_t* his_x   = (bf16_t*)alloc((size_t)BATCH * HISL * 256 * 2);
    bf16_t* qbuf    = (bf16_t*)alloc((size_t)BATCH * NH * SEQL * DK * 2);
    bf16_t* kbuf    = (bf16_t*)alloc((size_t)BATCH * NH * SEQL * DK * 2);
    bf16_t* vbuf    = (bf16_t*)alloc((size_t)BATCH * NH * SEQL * DK * 2);
    bf16_t* ctxb    = (bf16_t*)alloc((size_t)BATCH * SEQL * HID * 2);
    float*  poi_sum = (float*)alloc((size_t)BATCH * 256 * 4);
    float*  poi_hid = (float*)alloc((size_t)BATCH * 256 * 4);
    float*  aqb     = (float*)alloc((size_t)BATCH * 256 * 4);
    bf16_t* seqh    = (bf16_t*)alloc((size_t)BATCH * 512 * 2);
    (void)ws_size; (void)in_sizes; (void)n_in; (void)out_size;

    auto grid = [](size_t n, int b) { return (unsigned)((n + b - 1) / b); };

    // 1. pack/transpose weights to bf16 [K,N]
    pack_t<<<grid((size_t)256*128,256),256,0,stream>>>(main_W,  main_Wt, 256, 128);
    pack_t<<<grid((size_t)256*256,256),256,0,stream>>>(other_W, other_Wt,256, 256);
    pack_t<<<grid((size_t)256*256,256),256,0,stream>>>(sWq, sWq_t, 256, 256);
    pack_t<<<grid((size_t)256*256,256),256,0,stream>>>(sWk, sWk_t, 256, 256);
    pack_t<<<grid((size_t)256*256,256),256,0,stream>>>(sWv, sWv_t, 256, 256);
    pack_t<<<grid((size_t)256*256,256),256,0,stream>>>(sWo, sWo_t, 256, 256);
    pack_t<<<grid((size_t)256*256,256),256,0,stream>>>(aWk, aWk_t, 256, 256);
    pack_t<<<grid((size_t)256*256,256),256,0,stream>>>(aWv, aWv_t, 256, 256);
    pack_t<<<grid((size_t)128*POI,256),256,0,stream>>>(W_time, Wtime_t, 128, POI);
    pack_t<<<grid((size_t)128*POI,256),256,0,stream>>>(W_cat,  Wcat_t,  128, POI);
    pack_t<<<grid((size_t)POI*512,256),256,0,stream>>>(poi_W, poi_Wt, POI, 512);
    pack_t<<<grid((size_t)CATN*512,256),256,0,stream>>>(cat_W, cat_Wt, CATN, 512);

    // 2. time/cat tables
    tab_gemm<<<TVN/16, 32, 0, stream>>>(tp_one, Wtime_t, b_time, time_tab);
    tab_gemm<<<CATN/16, 32, 0, stream>>>(ct_one, Wcat_t, b_cat, cat_tab);

    // 3. embeddings
    embed_kernel<<<(BATCH*SEQL)/16, 64, 0, stream>>>(traj, timev, catv, poi_tab, poi_pin,
        time_tab, cat_tab, main_Wt, main_b, other_Wt, other_b, freq_W, seq_x, SEQL);
    embed_kernel<<<(BATCH*HISL)/16, 64, 0, stream>>>(his_traj, his_time, his_cat, poi_tab, poi_pin,
        time_tab, cat_tab, main_Wt, main_b, other_Wt, other_b, freq_W, his_x, HISL);

    // 4. QKV
    gemm_qkv<<<2048, 256, 0, stream>>>(seq_x, sWq_t, sbq, qbuf);
    gemm_qkv<<<2048, 256, 0, stream>>>(seq_x, sWk_t, sbk, kbuf);
    gemm_qkv<<<2048, 256, 0, stream>>>(seq_x, sWv_t, sbv, vbuf);

    // 5. fused attention (head0 probs -> d_out)
    size_t attn_smem = (size_t)64*256*2 + (size_t)4*16*256*2 + (size_t)256*4;
    attn_kernel<<<BATCH*NH, 128, attn_smem, stream>>>(qbuf, kbuf, vbuf, traj, ctxb, out_attn);

    // 6. output proj + LN + masked mean
    zero_f32<<<grid((size_t)BATCH*256,256),256,0,stream>>>(poi_sum, BATCH*256);
    swo_ln_kernel<<<(BATCH*SEQL)/64, 128, 0, stream>>>(ctxb, sWo_t, sbo, sn_g, sn_b, traj, poi_sum);
    finalize_kernel<<<BATCH, 256, 0, stream>>>(poi_sum, aWq, poi_hid, aqb, seqh);

    // 7. additive history attention -> his_hidden (second half of seq_hidden)
    addattn_kernel<<<BATCH, 256, 0, stream>>>(his_x, aWk_t, aWv_t, aWa, aWo, ln_g, ln_b,
                                              aqb, poi_hid, his_traj, seqh);

    // 8. logits
    logit_gemm<<<(16*(POI/16))/8, 256, 0, stream>>>(seqh, poi_Wt, poi_b, out_poi, POI/16);
    logit_gemm<<<(16*(CATN/16))/8, 256, 0, stream>>>(seqh, cat_Wt, cat_b, out_cat, CATN/16);

    // 9. top-10 category indices
    topk_cat<<<BATCH, 32, 0, stream>>>(out_cat, out_pred);
}
